// ExtractionNetwork_36627481101142
// MI455X (gfx1250) — matrix operand order repaired and checked by reading, not compiled, not run on hardware
//
#include <hip/hip_runtime.h>

// ---------------------------------------------------------------------------
// MoE extraction network for MI455X (gfx1250, wave32, WMMA).
// Core GEMMs: v_wmma_f32_16x16x32_f16, fused bias+LayerNorm+ReLU epilogue,
// direct packed-layout activation stores, and double-buffered ASYNC
// global->LDS staging of B tiles (global_load_async_to_lds_b128 + ASYNCcnt).
// ---------------------------------------------------------------------------

typedef __attribute__((ext_vector_type(16))) _Float16 v16h;
typedef __attribute__((ext_vector_type(8)))  _Float16 v8h;
typedef __attribute__((ext_vector_type(8)))  float    v8f;

#define BB 32768   // batch
#define DD 512     // input dim
#define HH 256     // hidden
#define OO 128     // expert out
#define EE 10      // experts
#define TT 3       // tasks
#define GG 64      // gate hidden

// Async copy of 16B per lane from global to LDS (no VGPR round-trip).
// VFLAT operand order: vdst (LDS byte address), vaddr (64b global), saddr.
__device__ __forceinline__ void async_copy_b128(unsigned lds_addr, const void* gptr) {
  asm volatile("global_load_async_to_lds_b128 %0, %1, off"
               :: "v"(lds_addr), "v"(gptr) : "memory");
}
__device__ __forceinline__ void wait_async0() {
  asm volatile("s_wait_asynccnt 0x0" ::: "memory");
}

// ---------------------------------------------------------------------------
// Pack an [R x K] row-major f32 matrix into WMMA A-fragment layout:
// tile(rt,kt) -> 32 lanes x 16 halves contiguous.  Per ISA 7.12.2 (16-bit A
// 16x32): lane = m + 16*h holds, at slot p:  K = (p<8) ? 8h+p : 16+8h+(p-8).
// ---------------------------------------------------------------------------
__global__ void pack_a_kernel(const float* __restrict__ src, _Float16* __restrict__ dst,
                              long total, int K) {
  long idx = (long)blockIdx.x * 256 + threadIdx.x;
  if (idx >= total) return;
  long tile = idx >> 9;            // 512 halves per 16x32 tile
  int  rem  = (int)(idx & 511);
  int  lane = rem >> 4;
  int  p    = rem & 15;
  int  KT   = K >> 5;
  long rt   = tile / KT;
  int  kt   = (int)(tile % KT);
  int  m    = lane & 15, h = lane >> 4;
  int  kk   = (p < 8) ? (8 * h + p) : (16 + 8 * h + (p - 8));
  long r    = rt * 16 + m;
  long k    = (long)kt * 32 + kk;
  dst[idx] = (_Float16)src[r * K + k];
}

// ---------------------------------------------------------------------------
// Pack weights W[e][K][N] (f32, row-major) into WMMA B-fragment layout:
// tile(nt,kt) -> 32 lanes x 16 halves; lane holds column N=lane%16 with
// K = kt*32 + ((p<8) ? 8h+p : 16+8h+(p-8)), h=lane/16 (matches A-fragment K order).
// ---------------------------------------------------------------------------
__global__ void pack_b_kernel(const float* __restrict__ W, _Float16* __restrict__ dst,
                              long total, int K, int N) {
  long idx = (long)blockIdx.x * 256 + threadIdx.x;
  if (idx >= total) return;
  long perE = (long)K * N;
  int  e    = (int)(idx / perE);
  long rem1 = idx % perE;
  long tile = rem1 >> 9;
  int  rem  = (int)(rem1 & 511);
  int  lane = rem >> 4;
  int  p    = rem & 15;
  int  KT   = K >> 5;
  int  nt   = (int)(tile / KT);
  int  kt   = (int)(tile % KT);
  int  n    = nt * 16 + (lane & 15);
  int  k    = kt * 32 + ((p < 8) ? (8 * (lane >> 4) + p) : (16 + 8 * (lane >> 4) + (p - 8)));
  dst[idx] = (_Float16)W[((long)e * K + k) * N + n];
}

// ---------------------------------------------------------------------------
// WMMA GEMM with fused bias (+ LayerNorm) + ReLU epilogue.
//   C[e] = act( A[e or shared] @ B[e] + bias[e] )
// Block = 256 threads (8 waves). Wave w owns a 16-row tile and the whole N
// extent (NT 16-col tiles -> per-row LN entirely in registers + shfl_xor).
// B tiles stream global->LDS via async loads, double-buffered: buffer for
// K-step kt+1 fills while WMMAs consume K-step kt. One barrier per K-step.
// OUT_PACKED: store directly in A-fragment layout for the next GEMM.
// ---------------------------------------------------------------------------
template <int NT, bool DO_LN, bool OUT_PACKED>
__global__ __launch_bounds__(256, 1)
void gemm_ln_relu_kernel(const _Float16* __restrict__ Apack, long aStride,
                         const _Float16* __restrict__ Bpack,
                         const float* __restrict__ bias,
                         const float* __restrict__ gamma,
                         const float* __restrict__ beta,
                         _Float16* __restrict__ Out,
                         int M, int KT) {
  const int N    = NT * 16;
  const int e    = blockIdx.y;
  const int wave = threadIdx.x >> 5;
  const int lane = threadIdx.x & 31;
  const long rt  = (long)blockIdx.x * 8 + wave;   // 16-row tile index in M

  __shared__ v8h lB[2 * NT * 64];                 // 2 buffers * NT tiles * 512 halves
  const unsigned ldsBase = (unsigned)(unsigned long long)&lB[0];

  const _Float16* A  = Apack + (long)e * aStride + rt * KT * 512;
  const _Float16* Be = Bpack + (long)e * NT * KT * 512;

  // Stage one K-step of B (NT tiles): 8 waves x NT/4 async b128 each.
  auto stage = [&](int kt, int buf) {
#pragma unroll
    for (int pss = 0; pss < NT / 4; ++pss) {
      int nt    = pss * 4 + (threadIdx.x >> 6);
      int chunk = threadIdx.x & 63;
      unsigned lds = ldsBase + (unsigned)(((buf * NT + nt) * 64 + chunk) * 16);
      async_copy_b128(lds, Be + ((long)(nt * KT + kt)) * 512 + chunk * 8);
    }
  };

  v8f acc[NT];
#pragma unroll
  for (int nt = 0; nt < NT; ++nt)
#pragma unroll
    for (int i = 0; i < 8; ++i) acc[nt][i] = 0.0f;

  stage(0, 0);                                    // prologue: fill buffer 0

#pragma unroll 1
  for (int kt = 0; kt < KT; ++kt) {
    const int cur = kt & 1;
    wait_async0();          // my buf[cur] async loads (issued last iter) landed
    __syncthreads();        // everyone's landed; everyone done reading buf[nxt]
    if (kt + 1 < KT) stage(kt + 1, cur ^ 1);      // overlaps with WMMAs below

    v16h a = *(const v16h*)(A + (long)kt * 512 + lane * 16);
#pragma unroll
    for (int nt = 0; nt < NT; ++nt) {
      v16h b = *(const v16h*)(lB + (cur * NT + nt) * 64 + lane * 2);
      acc[nt] = __builtin_amdgcn_wmma_f32_16x16x32_f16(
          false, a, false, b, (short)0, acc[nt], false, false);
    }
  }

  // ---- epilogue: bias (+ LN) + ReLU ----
  const int col  = lane & 15;
  const int half = lane >> 4;

  float biasv[NT], gv[NT], bv[NT];
#pragma unroll
  for (int nt = 0; nt < NT; ++nt) {
    biasv[nt] = bias[(long)e * N + nt * 16 + col];
    if (DO_LN) {
      gv[nt] = gamma[(long)e * N + nt * 16 + col];
      bv[nt] = beta [(long)e * N + nt * 16 + col];
    }
  }
#pragma unroll
  for (int nt = 0; nt < NT; ++nt)
#pragma unroll
    for (int v = 0; v < 8; ++v) acc[nt][v] += biasv[nt];

  const int  KTp         = N >> 5;                          // packed K-tiles of out
  const long rowBase     = (long)e * M + rt * 16;           // row-major base
  const long tileRowBase = ((long)e * (M >> 4) + rt) * KTp * 512;  // packed base

#pragma unroll
  for (int v = 0; v < 8; ++v) {
    float mean = 0.f, rstd = 1.f;
    if (DO_LN) {
      float s = 0.f, s2 = 0.f;
#pragma unroll
      for (int nt = 0; nt < NT; ++nt) {
        float x = acc[nt][v];
        s += x; s2 += x * x;
      }
      // reduce across the 16 lanes of this half-wave (C rows live per half)
#pragma unroll
      for (int off = 1; off <= 8; off <<= 1) {
        s  += __shfl_xor(s,  off, 32);
        s2 += __shfl_xor(s2, off, 32);
      }
      mean = s / N;
      float var = s2 / N - mean * mean;
      rstd = rsqrtf(var + 1e-5f);
    }
    const int m = v + 8 * half;                 // row within 16-row tile
#pragma unroll
    for (int nt = 0; nt < NT; ++nt) {
      float x = acc[nt][v];
      if (DO_LN) x = (x - mean) * rstd * gv[nt] + bv[nt];
      _Float16 hv = (_Float16)fmaxf(x, 0.f);
      if (OUT_PACKED) {
        // inverse A-fragment mapping for element (row m, col kk of ktp)
        int kk  = 16 * (nt & 1) + col;
        int hp  = (kk < 16) ? (kk >> 3) : ((kk - 16) >> 3);
        int p   = (kk < 16) ? (kk & 7) : (8 + (kk & 7));
        int ktp = nt >> 1;
        Out[tileRowBase + (long)ktp * 512 + (m + 16 * hp) * 16 + p] = hv;
      } else {
        Out[(rowBase + m) * N + nt * 16 + col] = hv;
      }
    }
  }
}

// ---------------------------------------------------------------------------
// Second gate layer + softmax: out[t][b][:] = softmax(gh[t][b][:] @ w2[t] + b2[t])
// G=64, Eo<=16 -> trivial VALU work, one thread per (t,b).
// ---------------------------------------------------------------------------
__global__ void gate2_softmax_kernel(const _Float16* __restrict__ gh,
                                     const float* __restrict__ w2,
                                     const float* __restrict__ b2,
                                     float* __restrict__ out,
                                     int T, int Bn, int G, int Eo) {
  long idx = (long)blockIdx.x * 256 + threadIdx.x;
  if (idx >= (long)T * Bn) return;
  int  t = (int)(idx / Bn);
  long b = idx % Bn;
  const _Float16* hrow = gh + ((long)t * Bn + b) * G;
  float acc[16];
  for (int j = 0; j < Eo; ++j) acc[j] = b2[t * Eo + j];
  for (int g = 0; g < G; ++g) {
    float hv = (float)hrow[g];
    const float* wr = w2 + ((long)t * G + g) * Eo;
    for (int j = 0; j < Eo; ++j) acc[j] += hv * wr[j];
  }
  float mx = acc[0];
  for (int j = 1; j < Eo; ++j) mx = fmaxf(mx, acc[j]);
  float s = 0.f;
  for (int j = 0; j < Eo; ++j) { acc[j] = __expf(acc[j] - mx); s += acc[j]; }
  float inv = 1.f / s;
  float* orow = out + ((long)t * Bn + b) * Eo;
  for (int j = 0; j < Eo; ++j) orow[j] = acc[j] * inv;
}

// ---------------------------------------------------------------------------
// Final gated mixtures.  d_out layout: [T=3 task_out ; 1 shared_out][B][O].
// combined experts for task t: {0,1,2,3, 4+2t, 5+2t}.
// ---------------------------------------------------------------------------
__global__ void combine_kernel(const _Float16* __restrict__ eo,  // [E][B][O] f16
                               const float* __restrict__ gw,     // [T][B][6]
                               const float* __restrict__ sgw,    // [B][E]
                               float* __restrict__ out, int Bn) {
  long idx = (long)blockIdx.x * 256 + threadIdx.x;
  if (idx >= (long)Bn * OO) return;
  long b = idx / OO;
  int  o = (int)(idx % OO);
  float ev[EE];
#pragma unroll
  for (int e = 0; e < EE; ++e)
    ev[e] = (float)eo[((long)e * Bn + b) * OO + o];
#pragma unroll
  for (int t = 0; t < TT; ++t) {
    const float* g = gw + ((long)t * Bn + b) * 6;
    float a = g[0] * ev[0] + g[1] * ev[1] + g[2] * ev[2] + g[3] * ev[3] +
              g[4] * ev[4 + 2 * t] + g[5] * ev[5 + 2 * t];
    out[((long)t * Bn + b) * OO + o] = a;
  }
  const float* s = sgw + b * EE;
  float a = 0.f;
#pragma unroll
  for (int e = 0; e < EE; ++e) a += s[e] * ev[e];
  out[((long)TT * Bn + b) * OO + o] = a;
}

// ---------------------------------------------------------------------------
extern "C" void kernel_launch(void* const* d_in, const int* in_sizes, int n_in,
                              void* d_out, int out_size, void* d_ws, size_t ws_size,
                              hipStream_t stream) {
  (void)in_sizes; (void)n_in; (void)out_size; (void)ws_size;
  const float* x     = (const float*)d_in[0];
  const float* ew1   = (const float*)d_in[1];
  const float* eb1   = (const float*)d_in[2];
  const float* eg1   = (const float*)d_in[3];
  const float* ebt1  = (const float*)d_in[4];
  const float* ew2   = (const float*)d_in[5];
  const float* eb2   = (const float*)d_in[6];
  const float* eg2   = (const float*)d_in[7];
  const float* ebt2  = (const float*)d_in[8];
  const float* ew3   = (const float*)d_in[9];
  const float* eb3   = (const float*)d_in[10];
  const float* eg3   = (const float*)d_in[11];
  const float* ebt3  = (const float*)d_in[12];
  const float* tg_w1 = (const float*)d_in[13];
  const float* tg_b1 = (const float*)d_in[14];
  const float* tg_w2 = (const float*)d_in[15];
  const float* tg_b2 = (const float*)d_in[16];
  const float* sg_w1 = (const float*)d_in[17];
  const float* sg_b1 = (const float*)d_in[18];
  const float* sg_w2 = (const float*)d_in[19];
  const float* sg_b2 = (const float*)d_in[20];

  // ---- workspace carve-up (256B aligned) ----
  char*  base = (char*)d_ws;
  size_t off  = 0;
  auto alloc = [&](size_t bytes) -> char* {
    char* p = base + off;
    off = (off + bytes + 255) & ~(size_t)255;
    return p;
  };
  _Float16* xpack  = (_Float16*)alloc((size_t)BB * DD * 2);          // 32 MB
  _Float16* ew1p   = (_Float16*)alloc((size_t)EE * DD * HH * 2);
  _Float16* ew2p   = (_Float16*)alloc((size_t)EE * HH * HH * 2);
  _Float16* ew3p   = (_Float16*)alloc((size_t)EE * HH * OO * 2);
  _Float16* tgw1p  = (_Float16*)alloc((size_t)TT * DD * GG * 2);
  _Float16* sgw1p  = (_Float16*)alloc((size_t)DD * GG * 2);
  _Float16* hpk1   = (_Float16*)alloc((size_t)EE * BB * HH * 2);     // 160 MB
  _Float16* hpk2   = (_Float16*)alloc((size_t)EE * BB * HH * 2);     // 160 MB
  _Float16* eo_rm  = (_Float16*)alloc((size_t)EE * BB * OO * 2);     // 80 MB
  _Float16* gh_rm  = (_Float16*)alloc((size_t)TT * BB * GG * 2);
  _Float16* sgh_rm = (_Float16*)alloc((size_t)BB * GG * 2);
  float*    gw     = (float*)   alloc((size_t)TT * BB * 6 * 4);
  float*    sgw    = (float*)   alloc((size_t)BB * EE * 4);

  const dim3 blk(256);
  auto nb = [](long total) { return (unsigned)((total + 255) / 256); };

  // ---- pack operands into WMMA fragment layouts ----
  pack_a_kernel<<<nb((long)BB * DD), blk, 0, stream>>>(x, xpack, (long)BB * DD, DD);
  pack_b_kernel<<<nb((long)EE * DD * HH), blk, 0, stream>>>(ew1,   ew1p,  (long)EE * DD * HH, DD, HH);
  pack_b_kernel<<<nb((long)EE * HH * HH), blk, 0, stream>>>(ew2,   ew2p,  (long)EE * HH * HH, HH, HH);
  pack_b_kernel<<<nb((long)EE * HH * OO), blk, 0, stream>>>(ew3,   ew3p,  (long)EE * HH * OO, HH, OO);
  pack_b_kernel<<<nb((long)TT * DD * GG), blk, 0, stream>>>(tg_w1, tgw1p, (long)TT * DD * GG, DD, GG);
  pack_b_kernel<<<nb((long)DD * GG),      blk, 0, stream>>>(sg_w1, sgw1p, (long)DD * GG,      DD, GG);

  const long hExpStride = (long)(BB / 16) * (HH / 32) * 512;  // per-expert A stride (halves)

  // ---- expert layer 1: [32768x512]@[512x256], LN+ReLU, packed f16 out ----
  gemm_ln_relu_kernel<16, true, true><<<dim3(BB / 128, EE), blk, 0, stream>>>(
      xpack, 0L, ew1p, eb1, eg1, ebt1, hpk1, BB, DD / 32);

  // ---- expert layer 2: [32768x256]@[256x256], LN+ReLU, packed f16 out ----
  gemm_ln_relu_kernel<16, true, true><<<dim3(BB / 128, EE), blk, 0, stream>>>(
      hpk1, hExpStride, ew2p, eb2, eg2, ebt2, hpk2, BB, HH / 32);

  // ---- expert layer 3: [32768x256]@[256x128], LN+ReLU -> eo (row-major) ----
  gemm_ln_relu_kernel<8, true, false><<<dim3(BB / 128, EE), blk, 0, stream>>>(
      hpk2, hExpStride, ew3p, eb3, eg3, ebt3, eo_rm, BB, HH / 32);

  // ---- gate first layers: [32768x512]@[512x64], bias+ReLU (no LN) ----
  gemm_ln_relu_kernel<4, false, false><<<dim3(BB / 128, TT), blk, 0, stream>>>(
      xpack, 0L, tgw1p, tg_b1, nullptr, nullptr, gh_rm, BB, DD / 32);
  gemm_ln_relu_kernel<4, false, false><<<dim3(BB / 128, 1), blk, 0, stream>>>(
      xpack, 0L, sgw1p, sg_b1, nullptr, nullptr, sgh_rm, BB, DD / 32);

  // ---- gate second layers + softmax ----
  gate2_softmax_kernel<<<nb((long)TT * BB), blk, 0, stream>>>(
      gh_rm, tg_w2, tg_b2, gw, TT, BB, GG, 6);
  gate2_softmax_kernel<<<nb((long)BB), blk, 0, stream>>>(
      sgh_rm, sg_w2, sg_b2, sgw, 1, BB, GG, EE);

  // ---- final gated mixtures -> d_out [4][B][O] f32 ----
  combine_kernel<<<nb((long)BB * OO), blk, 0, stream>>>(
      eo_rm, gw, sgw, (float*)d_out, BB);
}
